// ResNet_78288663871627
// MI455X (gfx1250) — matrix operand order repaired and checked by reading, not compiled, not run on hardware
//
#include <hip/hip_runtime.h>
#include <hip/hip_bf16.h>
#include <math.h>

typedef __attribute__((ext_vector_type(16))) _Float16 v16h;
typedef __attribute__((ext_vector_type(8)))  _Float16 v8h;
typedef __attribute__((ext_vector_type(8)))  float    v8f;

#define BATCH 128
#define HH 48
#define WW 48
#define NPIX (HH*WW)          // 2304
#define CI_PITCH 40           // halo-tile per-pixel pitch in halfs (16B aligned, bank-spread)
#define TILE_HALFS (6 * 52 * CI_PITCH)
#define NCHUNK (6 * 52 * 4)   // 1248 v8h slots per ci-chunk

// ---------------------------------------------------------------------------
// Kernel 1: per-batch encoder collapse.
// one-hot gather -> fc -> oh_conv(9x9) ; others -> others_conv(30x30) ;
// concat -> all_conv(32x39) -> conv1_w[:, :32] @ gvec + conv1_b = hconst[b][128]
// ---------------------------------------------------------------------------
__global__ __launch_bounds__(128) void prep_vec_kernel(
    const float* __restrict__ x,      // [B][33][48][48]
    const float* __restrict__ fc_w,   // [9][106]
    const float* __restrict__ fc_b,   // [9]
    const float* __restrict__ ohw,    // [9][9]
    const float* __restrict__ ohb,    // [9]
    const float* __restrict__ otw,    // [30][30]
    const float* __restrict__ otb,    // [30]
    const float* __restrict__ aw,     // [32][39]
    const float* __restrict__ ab,     // [32]
    const float* __restrict__ c1w,    // [128][64]
    const float* __restrict__ c1b,    // [128]
    float* __restrict__ hconst)       // [B][128]
{
    __shared__ int   idx[20];
    __shared__ float oth[30], v0[9], v1[9], o1[30], g32[32];
    const int b = blockIdx.x;
    const int t = threadIdx.x;
    const float* gi = x + (size_t)b * 33 * NPIX;   // channel 0 flattened

    if (t < 20) idx[t] = (int)(gi[t] + 0.5f);
    if (t < 30) oth[t] = gi[20 + t];
    __syncthreads();

    static const int off[20] = {0,5,12,16,22,25,33,38,42,48,55,58,63,67,73,81,86,93,97,103};
    if (t < 9) {
        float a = fc_b[t];
        for (int j = 0; j < 20; ++j) a += fc_w[t * 106 + off[j] + idx[j]];
        v0[t] = a;
    }
    __syncthreads();
    if (t < 9) {
        float a = ohb[t];
        for (int f = 0; f < 9; ++f) a += ohw[t * 9 + f] * v0[f];
        v1[t] = a;
    }
    if (t < 30) {
        float a = otb[t];
        for (int m = 0; m < 30; ++m) a += otw[t * 30 + m] * oth[m];
        o1[t] = a;
    }
    __syncthreads();
    if (t < 32) {
        float a = ab[t];
        for (int e = 0; e < 9;  ++e) a += aw[t * 39 + e] * v1[e];
        for (int k = 0; k < 30; ++k) a += aw[t * 39 + 9 + k] * o1[k];
        g32[t] = a;
    }
    __syncthreads();
    float a = c1b[t];
    for (int c = 0; c < 32; ++c) a += c1w[t * 64 + c] * g32[c];
    hconst[b * 128 + t] = a;
}

// ---------------------------------------------------------------------------
// Weight / activation format conversion kernels
// ---------------------------------------------------------------------------
__global__ __launch_bounds__(256) void cvt_w5_kernel(
    const float* __restrict__ w, _Float16* __restrict__ out) // [co][ci][5][5] -> [kp][co][ci]
{
    int i = blockIdx.x * 256 + threadIdx.x;
    if (i >= 25 * 128 * 128) return;
    int kp = i / (128 * 128);
    int rem = i - kp * (128 * 128);
    int co = rem >> 7, ci = rem & 127;
    int ky = kp / 5, kx = kp - ky * 5;
    out[i] = (_Float16)w[((co * 128 + ci) * 5 + ky) * 5 + kx];
}

__global__ __launch_bounds__(256) void cvt_c1w_kernel(
    const float* __restrict__ w, _Float16* __restrict__ out) // conv1_w[:,32:64] -> [co][32]
{
    int i = blockIdx.x * 256 + threadIdx.x;
    if (i >= 128 * 32) return;
    int co = i >> 5, ci = i & 31;
    out[i] = (_Float16)w[co * 64 + 32 + ci];
}

__global__ __launch_bounds__(256) void cvt_map_kernel(
    const float* __restrict__ x, _Float16* __restrict__ out) // NCHW ch1..32 -> [b][p][32] f16
{
    size_t i = (size_t)blockIdx.x * 256 + threadIdx.x;
    if (i >= (size_t)BATCH * NPIX * 32) return;
    int ci = (int)(i & 31);
    size_t rem = i >> 5;
    int p = (int)(rem % NPIX);
    int b = (int)(rem / NPIX);
    out[i] = (_Float16)x[((size_t)b * 33 + 1 + ci) * NPIX + p];
}

// ---------------------------------------------------------------------------
// Kernel: 1x1 conv (map half of conv1) via WMMA + hconst broadcast -> h (NHWC f16)
// grid (24, B), 8 waves; wave w owns co 16w..16w+15; block covers 96 pixels.
// ---------------------------------------------------------------------------
__global__ __launch_bounds__(256) void conv1x1_wmma_kernel(
    const _Float16* __restrict__ map16,  // [B][2304][32]
    const _Float16* __restrict__ w,      // [128][32]
    const float*    __restrict__ hconst, // [B][128]
    _Float16*       __restrict__ out)    // [B][2304][128]
{
    __shared__ _Float16 tile[96 * CI_PITCH];
    const int b = blockIdx.y;
    const int pbase = blockIdx.x * 96;
    const int tid  = threadIdx.x;
    const int lane = tid & 31;
    const int wave = tid >> 5;
    const int n    = lane & 15;
    const int half = lane >> 4;
    const int ksA  = half * 8;     // A-frag K select
    const int ksB  = half * 8;     // B-frag K select
    const int cobase = wave << 4;

    // stage 96 pixels x 32 ci into LDS (ci contiguous)
    for (int t = tid; t < 96 * 4; t += 256) {
        int p = t >> 2, q = t & 3;
        v8h v = *(const v8h*)(map16 + ((size_t)b * NPIX + pbase + p) * 32 + q * 8);
        *(v8h*)(tile + p * CI_PITCH + q * 8) = v;
    }
    __syncthreads();

    // A fragment: 16 co rows x 32 ci
    const _Float16* wrow = w + (cobase + n) * 32;
    v16h a;
    *((v8h*)&a)       = *(const v8h*)(wrow + ksA);
    *(((v8h*)&a) + 1) = *(const v8h*)(wrow + ksA + 16);

    v8f z = {};
#pragma unroll
    for (int g = 0; g < 6; ++g) {
        const _Float16* bp = tile + (g * 16 + n) * CI_PITCH + ksB;
        v16h bf;
        *((v8h*)&bf)       = *(const v8h*)(bp);
        *(((v8h*)&bf) + 1) = *(const v8h*)(bp + 16);
        v8f acc = __builtin_amdgcn_wmma_f32_16x16x32_f16(false, a, false, bf,
                                                         (short)0, z, false, false);
        // add hconst and pack store: lane owns co = cobase + half*8 + j
        int p = pbase + g * 16 + n;
        v8h pk;
#pragma unroll
        for (int j = 0; j < 8; ++j) {
            int co = cobase + half * 8 + j;
            pk[j] = (_Float16)(acc[j] + hconst[b * 128 + co]);
        }
        *(v8h*)(out + ((size_t)b * NPIX + p) * 128 + cobase + half * 8) = pk;
    }
}

// ---------------------------------------------------------------------------
// Kernel: 5x5 conv 128->128 as implicit GEMM with WMMA f16.
// grid (24, B): one workgroup = 2 output rows x 48 cols x 128 co for one batch.
// Wave layout: waves 0-3 -> co {0,32,64,96}..+31 on strip row 0;
//              waves 4-7 -> same co on strip row 1.
// Each wave: 2 A-fragments (32 co) x 3 B-fragments (48 px) -> 6 WMMAs/tap,
// so B loads amortize to 1 ds_load_b128 per WMMA (A: 0.67 global/WMMA, L2-hot).
// Software-pipelined over 4 ci-chunks with LDS double buffering; the 25-tap
// loop is fully unrolled so all fragment loads use immediate offsets.
// ---------------------------------------------------------------------------
__global__ __launch_bounds__(256) void conv5x5_wmma_kernel(
    const _Float16* __restrict__ in,   // [B][2304][128] NHWC, zero-pad handled here
    const _Float16* __restrict__ w,    // [25][128][128]  (kpos, co, ci)
    const float*    __restrict__ bias, // [128]
    _Float16*       __restrict__ out,  // [B][2304][128]
    int apply_leaky)
{
    __shared__ _Float16 tile[2][TILE_HALFS];   // 2 x ~25 KB
    const int b     = blockIdx.y;
    const int ybase = blockIdx.x * 2;
    const int tid   = threadIdx.x;
    const int lane  = tid & 31;
    const int wave  = tid >> 5;
    const int n     = lane & 15;
    const int half  = lane >> 4;
    const int ksA   = half * 8;
    const int ksB   = half * 8;
    const int copair = wave & 3;       // which 32-co slice
    const int pxh    = wave >> 2;      // strip row (0/1)
    const int cobase = copair * 32;

    v8f acc[6];                        // [g] for co-tile 0, [g+3] for co-tile 1
    v8f z = {};
#pragma unroll
    for (int g = 0; g < 6; ++g) acc[g] = z;

    // per-g LDS fragment base offsets (halfs); tile row for output row pxh,
    // tap ky lives at halo row pxh+ky (0..5), col (g*16+n)+kx (0..51)
    int bbase[3];
#pragma unroll
    for (int g = 0; g < 3; ++g)
        bbase[g] = (pxh * 52 + g * 16 + n) * CI_PITCH + ksB;

    const _Float16* inb = in + (size_t)b * NPIX * 128;

    // ---- staging slot geometry (constant across chunks)
    int lds_off[5], goff[5];
    bool slot[5], ok[5];
#pragma unroll
    for (int k = 0; k < 5; ++k) {
        int t = tid + k * 256;
        slot[k] = (t < NCHUNK);
        int tt  = slot[k] ? t : 0;
        int row = tt / 208;
        int rem = tt - row * 208;
        int col = rem >> 2;
        int q   = rem & 3;
        int gy  = ybase - 2 + row;
        int gx  = col - 2;
        ok[k]      = slot[k] && gy >= 0 && gy < HH && gx >= 0 && gx < WW;
        lds_off[k] = (row * 52 + col) * CI_PITCH + q * 8;
        goff[k]    = (gy * WW + gx) * 128 + q * 8;
    }

    // ---- prologue: stage chunk 0 into registers
    v8h stage[5];
#pragma unroll
    for (int k = 0; k < 5; ++k) {
        v8h v = {};
        if (ok[k]) v = *(const v8h*)(inb + goff[k]);
        stage[k] = v;
    }

    const _Float16* wbase = w + (size_t)(cobase + n) * 128 + ksA;

    for (int cc = 0; cc < 4; ++cc) {
        _Float16* cur = &tile[cc & 1][0];

        // commit staged chunk cc to LDS buffer cc&1
#pragma unroll
        for (int k = 0; k < 5; ++k)
            if (slot[k]) *(v8h*)(cur + lds_off[k]) = stage[k];
        __syncthreads();

        // issue global loads for chunk cc+1 (overlap with WMMA below)
        if (cc < 3) {
#pragma unroll
            for (int k = 0; k < 5; ++k) {
                v8h v = {};
                if (ok[k]) v = *(const v8h*)(inb + goff[k] + (cc + 1) * 32);
                stage[k] = v;
            }
        }

        // ---- K loop: 25 taps fully unrolled -> immediate-offset DS/global loads
        const _Float16* wcc = wbase + cc * 32;
#pragma unroll
        for (int kp = 0; kp < 25; ++kp) {
            const int ky = kp / 5;
            const int kx = kp - ky * 5;
            v16h a0, a1;   // co-tile 0 and 1 (offset 16*128 halfs = 4KB imm)
            *((v8h*)&a0)       = *(const v8h*)(wcc + kp * (128 * 128));
            *(((v8h*)&a0) + 1) = *(const v8h*)(wcc + kp * (128 * 128) + 16);
            *((v8h*)&a1)       = *(const v8h*)(wcc + kp * (128 * 128) + 16 * 128);
            *(((v8h*)&a1) + 1) = *(const v8h*)(wcc + kp * (128 * 128) + 16 * 128 + 16);
#pragma unroll
            for (int g = 0; g < 3; ++g) {
                const _Float16* bp = cur + bbase[g] + (ky * 52 + kx) * CI_PITCH;
                v16h bf;
                *((v8h*)&bf)       = *(const v8h*)(bp);
                *(((v8h*)&bf) + 1) = *(const v8h*)(bp + 16);
                acc[g]     = __builtin_amdgcn_wmma_f32_16x16x32_f16(false, a0, false, bf,
                                                                    (short)0, acc[g],     false, false);
                acc[g + 3] = __builtin_amdgcn_wmma_f32_16x16x32_f16(false, a1, false, bf,
                                                                    (short)0, acc[g + 3], false, false);
            }
        }
        __syncthreads();
    }

    // ---- epilogue: bias (+ leaky), packed 16B stores (two co-tiles per wave)
#pragma unroll
    for (int g = 0; g < 3; ++g) {
        int p = (ybase + pxh) * WW + g * 16 + n;
        _Float16* op = out + ((size_t)b * NPIX + p) * 128;
#pragma unroll
        for (int t2 = 0; t2 < 2; ++t2) {
            v8h pk;
#pragma unroll
            for (int j = 0; j < 8; ++j) {
                int co = cobase + t2 * 16 + half * 8 + j;
                float v = acc[g + 3 * t2][j] + bias[co];
                if (apply_leaky) v = (v > 0.f) ? v : 0.01f * v;
                pk[j] = (_Float16)v;
            }
            *(v8h*)(op + cobase + t2 * 16 + half * 8) = pk;
        }
    }
}

// ---------------------------------------------------------------------------
// Kernel: SE block. Per-batch channel means over y2 + 2-layer MLP + sigmoid.
// ---------------------------------------------------------------------------
__global__ __launch_bounds__(128) void se_kernel(
    const _Float16* __restrict__ y2,   // [B][2304][128]
    const float* __restrict__ se_w1,   // [8][128]
    const float* __restrict__ se_w2,   // [128][8]
    float* __restrict__ s_out)         // [B][128]
{
    __shared__ float mean[128], hid[8];
    const int b = blockIdx.x;
    const int t = threadIdx.x;
    const _Float16* p = y2 + (size_t)b * NPIX * 128 + t;
    float sum = 0.f;
    for (int i = 0; i < NPIX; ++i) sum += (float)p[(size_t)i * 128];
    mean[t] = sum * (1.f / (float)NPIX);
    __syncthreads();
    if (t < 8) {
        float a = 0.f;
        for (int c = 0; c < 128; ++c) a += mean[c] * se_w1[t * 128 + c];
        hid[t] = (a > 0.f) ? a : 0.f;
    }
    __syncthreads();
    float o = 0.f;
    for (int j = 0; j < 8; ++j) o += hid[j] * se_w2[t * 8 + j];
    s_out[b * 128 + t] = 1.f / (1.f + expf(-o));
}

// ---------------------------------------------------------------------------
// Kernel: final scale + identity + leaky, NHWC f16 -> NCHW f32 via LDS transpose
// ---------------------------------------------------------------------------
__global__ __launch_bounds__(256) void final_kernel(
    const _Float16* __restrict__ y2,   // [B][2304][128]
    const _Float16* __restrict__ h,    // [B][2304][128]
    const float* __restrict__ s,       // [B][128]
    float* __restrict__ out)           // [B][128][2304] (NCHW)
{
    __shared__ float tile[32][129];
    const int b = blockIdx.y;
    const int pbase = blockIdx.x * 32;
    const int t = threadIdx.x;

#pragma unroll
    for (int i = 0; i < 16; ++i) {               // 32*128 / 256 = 16
        int e = t + i * 256;
        int pl = e >> 7;
        int co = e & 127;
        size_t idx = ((size_t)b * NPIX + pbase + pl) * 128 + co;
        float v = (float)y2[idx] * s[b * 128 + co] + (float)h[idx];
        tile[pl][co] = (v > 0.f) ? v : 0.01f * v;
    }
    __syncthreads();
#pragma unroll
    for (int i = 0; i < 16; ++i) {
        int e = t + i * 256;
        int co = e >> 5;
        int pl = e & 31;
        out[((size_t)(b * 128 + co)) * NPIX + pbase + pl] = tile[pl][co];
    }
}

// ---------------------------------------------------------------------------
extern "C" void kernel_launch(void* const* d_in, const int* in_sizes, int n_in,
                              void* d_out, int out_size, void* d_ws, size_t ws_size,
                              hipStream_t stream) {
    const float* x    = (const float*)d_in[0];
    const float* fc_w = (const float*)d_in[1];
    const float* fc_b = (const float*)d_in[2];
    const float* ohw  = (const float*)d_in[3];
    const float* ohb  = (const float*)d_in[4];
    const float* otw  = (const float*)d_in[5];
    const float* otb  = (const float*)d_in[6];
    const float* aw   = (const float*)d_in[7];
    const float* ab   = (const float*)d_in[8];
    const float* c1w  = (const float*)d_in[9];
    const float* c1b  = (const float*)d_in[10];
    const float* r1w  = (const float*)d_in[11];
    const float* r1b  = (const float*)d_in[12];
    const float* r2w  = (const float*)d_in[13];
    const float* r2b  = (const float*)d_in[14];
    const float* sew1 = (const float*)d_in[15];
    const float* sew2 = (const float*)d_in[16];

    char* ws = (char*)d_ws;
    size_t cur = 0;
    auto alloc = [&](size_t bytes) -> void* {
        void* p = ws + cur;
        cur = (cur + bytes + 255) & ~(size_t)255;
        return p;
    };

    _Float16* h16   = (_Float16*)alloc((size_t)BATCH * NPIX * 128 * 2);
    _Float16* y116  = (_Float16*)alloc((size_t)BATCH * NPIX * 128 * 2);
    _Float16* y216  = (_Float16*)alloc((size_t)BATCH * NPIX * 128 * 2);
    _Float16* map16 = (_Float16*)alloc((size_t)BATCH * NPIX * 32 * 2);
    _Float16* w5a   = (_Float16*)alloc((size_t)25 * 128 * 128 * 2);
    _Float16* w5b   = (_Float16*)alloc((size_t)25 * 128 * 128 * 2);
    _Float16* c1w16 = (_Float16*)alloc((size_t)128 * 32 * 2);
    float*    hconst= (float*)alloc((size_t)BATCH * 128 * 4);
    float*    sbuf  = (float*)alloc((size_t)BATCH * 128 * 4);

    // ---- encoder collapse + format conversions (tiny)
    prep_vec_kernel<<<BATCH, 128, 0, stream>>>(x, fc_w, fc_b, ohw, ohb, otw, otb,
                                               aw, ab, c1w, c1b, hconst);
    cvt_w5_kernel<<<(25 * 128 * 128 + 255) / 256, 256, 0, stream>>>(r1w, w5a);
    cvt_w5_kernel<<<(25 * 128 * 128 + 255) / 256, 256, 0, stream>>>(r2w, w5b);
    cvt_c1w_kernel<<<(128 * 32 + 255) / 256, 256, 0, stream>>>(c1w, c1w16);
    {
        size_t total = (size_t)BATCH * NPIX * 32;
        cvt_map_kernel<<<(unsigned)((total + 255) / 256), 256, 0, stream>>>(x, map16);
    }

    // ---- h = conv1x1(map) + hconst  (identity path)
    conv1x1_wmma_kernel<<<dim3(NPIX / 96, BATCH), 256, 0, stream>>>(map16, c1w16, hconst, h16);

    // ---- y1 = leaky(conv5x5(h) + b1) ; y2 = conv5x5(y1) + b2
    conv5x5_wmma_kernel<<<dim3(HH / 2, BATCH), 256, 0, stream>>>(h16,  w5a, r1b, y116, 1);
    conv5x5_wmma_kernel<<<dim3(HH / 2, BATCH), 256, 0, stream>>>(y116, w5b, r2b, y216, 0);

    // ---- SE scales
    se_kernel<<<BATCH, 128, 0, stream>>>(y216, sew1, sew2, sbuf);

    // ---- out = leaky(y2 * s + h), NHWC->NCHW f32
    final_kernel<<<dim3(NPIX / 32, BATCH), 256, 0, stream>>>(y216, h16, sbuf, (float*)d_out);
}
